// Net_32504312496632
// MI455X (gfx1250) — matrix operand order repaired and checked by reading, not compiled, not run on hardware
//
#include <hip/hip_runtime.h>
#include <hip/hip_bf16.h>

typedef __attribute__((ext_vector_type(16))) _Float16 v16h;
typedef __attribute__((ext_vector_type(8)))  _Float16 v8h;
typedef __attribute__((ext_vector_type(8)))  float    v8f;

#define NPAD 64          // 33 padded to 64 (4x4 WMMA tiles, K = 2 chunks of 32)
#define N1   33
#define NN1  (N1*N1)     // 1089

struct SM {
    float Dmat[NN1];                 // 33x33 node-cost matrix
    float S[N1*34];                  // sinkhorn working matrix (stride 34)
    alignas(32) _Float16 V16[NPAD*NPAD];   // V   (f16, padded)
    alignas(32) _Float16 Vt16[NPAD*NPAD];  // V^T (f16, padded)
    alignas(32) _Float16 Bm[NPAD*NPAD];    // CA2_p (f16, padded)
    alignas(32) _Float16 T16[NPAD*NPAD];   // T = V @ CA2_p (converted to f16)
    float Wb[NPAD*NPAD];             // W = T @ V^T (f32)
    float ncs[64];                   // node_costs 8x8
    float Cost5[25];                 // 5x5 edge-cost table
    float rowf[N1], colf[N1];
    int   l1s[32], l2s[32];
    unsigned char a1s[NN1], a2s[NN1];
    float qsum, lsum, nodeInsDel;
};

// A-fragment (16x32 f16): lane = (Khalf<<4)|M ; elems 0..7 -> K=k0+h*8.. ; elems 8..15 -> K=k0+16+h*8..
__device__ inline v16h load_afrag(const _Float16* M, int mr, int k0, int lane) {
    const int h = lane >> 4, m = lane & 15;
    const _Float16* row = M + (mr + m) * NPAD + k0;
    v8h lo = *(const v8h*)(row + h * 8);
    v8h hi = *(const v8h*)(row + 16 + h * 8);
    v16h a;
#pragma unroll
    for (int e = 0; e < 8; ++e) { a[e] = lo[e]; a[e + 8] = hi[e]; }
    return a;
}

// B-fragment (32x16 f16): lane = (Khalf<<4)|N ; elems 0..7 -> K=k0+h*8.. ; elems 8..15 -> K=k0+16+h*8..
__device__ inline v16h load_bfrag(const _Float16* M, int k0, int nc, int lane) {
    const int h = lane >> 4, n = lane & 15;
    const _Float16* col = M + k0 * NPAD + nc + n;
    v16h b;
#pragma unroll
    for (int e = 0; e < 8; ++e) { b[e] = col[(h * 8 + e) * NPAD]; b[e + 8] = col[(16 + h * 8 + e) * NPAD]; }
    return b;
}

// one 16x16 output tile of a 64x64x64 GEMM, accumulate f32
__device__ inline v8f gemm_tile(const _Float16* A, const _Float16* B, int MT, int NT, int lane) {
    v8f acc = {0.f, 0.f, 0.f, 0.f, 0.f, 0.f, 0.f, 0.f};
#pragma unroll
    for (int kc = 0; kc < 2; ++kc) {
        v16h a = load_afrag(A, MT * 16, kc * 32, lane);
        v16h b = load_bfrag(B, kc * 32, NT * 16, lane);
        acc = __builtin_amdgcn_wmma_f32_16x16x32_f16(false, a, false, b, (short)0, acc, false, false);
    }
    return acc;
}

__global__ __launch_bounds__(256)
void ged_kernel(const float* __restrict__ nw, const float* __restrict__ ew,
                const int* __restrict__ A, const int* __restrict__ labels,
                const int* __restrict__ pairs, float* __restrict__ ged) {
    __shared__ SM sm;
    const int tid  = threadIdx.x;
    const int lane = tid & 31;
    // wave index as an SGPR: tid>>5 is wave-uniform; readfirstlane makes that
    // visible to the compiler -> uniform control flow around WMMA (EXEC all-ones).
    const int wave = __builtin_amdgcn_readfirstlane(tid >> 5);
    const int pair = blockIdx.x;
    const int g1 = pairs[2 * pair + 0];
    const int g2 = pairs[2 * pair + 1];

    if (tid < 32) { sm.l1s[tid] = labels[g1 * 32 + tid]; sm.l2s[tid] = labels[g2 * 32 + tid]; }
    for (int idx = tid; idx < NN1; idx += 256) {
        int i = idx / N1, j = idx % N1;
        sm.a1s[idx] = (i < 32 && j < 32) ? (unsigned char)A[g1 * 1024 + i * 32 + j] : 0;
        sm.a2s[idx] = (i < 32 && j < 32) ? (unsigned char)A[g2 * 1024 + i * 32 + j] : 0;
    }
    if (tid == 0) {
        // node costs: relu(nw) into upper triangle (k=1), symmetrized
        for (int r = 0; r < 64; ++r) sm.ncs[r] = 0.f;
        int idx = 0;
        for (int r = 0; r < 8; ++r)
            for (int c = r + 1; c < 8; ++c, ++idx) {
                float v = fmaxf(nw[idx], 0.f);
                sm.ncs[r * 8 + c] = v; sm.ncs[c * 8 + r] = v;
            }
        sm.nodeInsDel = fmaxf(nw[28], 0.f);
        // edge costs 4x4 + Cost5 table: Cost5[p][q] = cost of edge-pair labels (0 = no edge)
        float ec[16];
        for (int r = 0; r < 16; ++r) ec[r] = 0.f;
        idx = 0;
        for (int r = 0; r < 4; ++r)
            for (int c = r + 1; c < 4; ++c, ++idx) {
                float v = fmaxf(ew[idx], 0.f);
                ec[r * 4 + c] = v; ec[c * 4 + r] = v;
            }
        float eid = fmaxf(ew[6], 0.f);
        for (int p = 0; p < 5; ++p)
            for (int q = 0; q < 5; ++q) {
                float v;
                if (p == 0 && q == 0) v = 0.f;
                else if (p == 0 || q == 0) v = eid;
                else v = ec[(p - 1) * 4 + (q - 1)];
                sm.Cost5[p * 5 + q] = v;
            }
        sm.qsum = 0.f; sm.lsum = 0.f;
    }
    __syncthreads();

    // Dmat + initial sinkhorn matrix S = exp(-0.5*Dmat)
    for (int idx = tid; idx < NN1; idx += 256) {
        int i = idx / N1, j = idx % N1;
        float d;
        if (i < 32 && j < 32)        d = sm.ncs[sm.l1s[i] * 8 + sm.l2s[j]];
        else if (i == 32 && j == 32) d = 0.f;
        else                         d = sm.nodeInsDel;
        sm.Dmat[idx] = d;
        sm.S[i * 34 + j] = expf(-0.5f * d);
    }
    __syncthreads();

    // 10 sinkhorn iterations (last row/col scale factor forced to 1)
    for (int it = 0; it < 10; ++it) {
        if (tid < N1) {
            float s = 0.f;
            for (int j = 0; j < N1; ++j) s += sm.S[tid * 34 + j];
            sm.rowf[tid] = (tid == 32) ? 1.f : (1.f / s);
        }
        __syncthreads();
        for (int idx = tid; idx < NN1; idx += 256) {
            int i = idx / N1, j = idx % N1;
            sm.S[i * 34 + j] *= sm.rowf[i];
        }
        __syncthreads();
        if (tid < N1) {
            float s = 0.f;
            for (int i = 0; i < N1; ++i) s += sm.S[i * 34 + tid];
            sm.colf[tid] = (tid == 32) ? 1.f : (1.f / s);
        }
        __syncthreads();
        for (int idx = tid; idx < NN1; idx += 256) {
            int i = idx / N1, j = idx % N1;
            sm.S[i * 34 + j] *= sm.colf[j];
        }
        __syncthreads();
    }

    // V (f16, zero-padded to 64) + V^T ; linear term lsum = sum(Dmat * V)
    for (int idx = tid; idx < NPAD * NPAD; idx += 256) {
        int i = idx >> 6, j = idx & 63;
        float v = (i < N1 && j < N1) ? sm.S[i * 34 + j] : 0.f;
        sm.V16[idx] = (_Float16)v;
        sm.Vt16[j * NPAD + i] = (_Float16)v;
    }
    {
        float part = 0.f;
        for (int idx = tid; idx < NN1; idx += 256) {
            int i = idx / N1, j = idx % N1;
            part += sm.Dmat[idx] * sm.S[i * 34 + j];
        }
        atomicAdd(&sm.lsum, part);
    }
    __syncthreads();

    // Q = sum_p sum_{A1[i,j]==p} (V @ CA2_p @ V^T)[j,i]
    for (int p = 0; p < 5; ++p) {
        // CA2_p[k][l] = Cost5[p][A2[k,l]], zero-padded
        for (int idx = tid; idx < NPAD * NPAD; idx += 256) {
            int k = idx >> 6, l = idx & 63;
            float c = (k < N1 && l < N1) ? sm.Cost5[p * 5 + sm.a2s[k * N1 + l]] : 0.f;
            sm.Bm[idx] = (_Float16)c;
        }
        __syncthreads();
        // T = V @ CA2_p : 8 waves x 2 tiles, fully unrolled, scalar tile index
        {
            const int h = lane >> 4, n = lane & 15;
#pragma unroll
            for (int tt = 0; tt < 2; ++tt) {
                const int t  = wave + tt * 8;      // SGPR
                const int MT = t >> 2, NT = t & 3; // SGPRs
                v8f acc = gemm_tile(sm.V16, sm.Bm, MT, NT, lane);
#pragma unroll
                for (int r = 0; r < 8; ++r)
                    sm.T16[(MT * 16 + r + 8 * h) * NPAD + NT * 16 + n] = (_Float16)acc[r];
            }
        }
        __syncthreads();
        // W = T @ V^T
        {
            const int h = lane >> 4, n = lane & 15;
#pragma unroll
            for (int tt = 0; tt < 2; ++tt) {
                const int t  = wave + tt * 8;
                const int MT = t >> 2, NT = t & 3;
                v8f acc = gemm_tile(sm.T16, sm.Vt16, MT, NT, lane);
#pragma unroll
                for (int r = 0; r < 8; ++r)
                    sm.Wb[(MT * 16 + r + 8 * h) * NPAD + NT * 16 + n] = acc[r];
            }
        }
        __syncthreads();
        // accumulate Q contribution for this p
        float part = 0.f;
        for (int idx = tid; idx < NN1; idx += 256) {
            int i = idx / N1, j = idx % N1;
            part += (sm.a1s[idx] == (unsigned char)p) ? sm.Wb[j * NPAD + i] : 0.f;
        }
        atomicAdd(&sm.qsum, part);
        __syncthreads();
    }

    if (tid == 0) ged[pair] = 0.5f * sm.qsum + sm.lsum;
}

__global__ void norm_kernel(const float* __restrict__ g, float* __restrict__ out, int P) {
    __shared__ float mn, mx;
    if (threadIdx.x == 0) {
        float a = g[0], b = g[0];
        for (int i = 1; i < P; ++i) { float v = g[i]; a = fminf(a, v); b = fmaxf(b, v); }
        mn = a; mx = b;
    }
    __syncthreads();
    for (int i = threadIdx.x; i < P; i += blockDim.x)
        out[i] = (g[i] - mn) / (mx - mn);
}

extern "C" void kernel_launch(void* const* d_in, const int* in_sizes, int n_in,
                              void* d_out, int out_size, void* d_ws, size_t ws_size,
                              hipStream_t stream) {
    (void)in_sizes; (void)n_in; (void)ws_size;
    const float* nw     = (const float*)d_in[0];
    const float* ew     = (const float*)d_in[1];
    const int*   A      = (const int*)d_in[2];
    const int*   labels = (const int*)d_in[3];
    const int*   pairs  = (const int*)d_in[4];
    float* out    = (float*)d_out;
    float* ged_ws = (float*)d_ws;
    const int P = out_size;   // 64 pairs

    ged_kernel<<<dim3(P), dim3(256), 0, stream>>>(nw, ew, A, labels, pairs, ged_ws);
    norm_kernel<<<dim3(1), dim3(64), 0, stream>>>(ged_ws, out, P);
}
